// RecSys_83391085019409
// MI455X (gfx1250) — matrix operand order repaired and checked by reading, not compiled, not run on hardware
//
#include <hip/hip_runtime.h>
#include <hip/hip_bf16.h>
#include <math.h>

// Problem constants (from reference)
#define B_   4096
#define T_   50
#define D_   256   // input dim = 2*H
#define H_   128
#define NG_  512   // 4*H gates
#define C_   128

// ---------- WMMA types / helpers ----------
typedef __attribute__((ext_vector_type(16))) __bf16        v16bf;
typedef __attribute__((ext_vector_type(16))) unsigned short v16u;
typedef __attribute__((ext_vector_type(8)))  float          v8f;
typedef __attribute__((ext_vector_type(4)))  unsigned int   u32x4;

union V16 { v16u v; u32x4 q[2]; };

// Load 16 bf16 halves as two 16B chunks (p0: halves 0..7, p1: halves 8..15)
__device__ __forceinline__ v16u ld2x16(const unsigned short* p0, const unsigned short* p1) {
  V16 r;
  r.q[0] = *(const u32x4*)p0;
  r.q[1] = *(const u32x4*)p1;
  return r.v;
}

__device__ __forceinline__ v8f wmma_bf(v16u a, v16u b, v8f c) {
  return __builtin_amdgcn_wmma_f32_16x16x32_bf16(
      false, __builtin_bit_cast(v16bf, a),
      false, __builtin_bit_cast(v16bf, b),
      (short)0, c, false, false);
}

__device__ __forceinline__ unsigned short f2bf(float f) {
  unsigned u = __builtin_bit_cast(unsigned, f);
  u += 0x7fffu + ((u >> 16) & 1u);           // round-to-nearest-even
  return (unsigned short)(u >> 16);
}

__device__ __forceinline__ float sigm(float x) { return 1.f / (1.f + __expf(-x)); }

// ---------- utility kernels ----------
__global__ void cvt_bf16(const float* __restrict__ src, unsigned short* __restrict__ dst, int n) {
  for (int i = blockIdx.x * blockDim.x + threadIdx.x; i < n; i += gridDim.x * blockDim.x)
    dst[i] = f2bf(src[i]);
}

__global__ void zero_f32(float* __restrict__ p, int n) {
  for (int i = blockIdx.x * blockDim.x + threadIdx.x; i < n; i += gridDim.x * blockDim.x)
    p[i] = 0.f;
}

// ---------- fused BiLSTM recurrence (WMMA) ----------
// grid.x = B/16 (16 batch rows per block), grid.y = 4: {q-fwd, q-bwd, qp-fwd, qp-bwd}
// 256 threads = 8 waves; each wave owns 4 of the 32 gate N-tiles (N=512).
__global__ __launch_bounds__(256) void bilstm_rec(
    const unsigned short* __restrict__ xq,    // [B,T,D] bf16
    const unsigned short* __restrict__ xqp,   // [B,T,D] bf16
    const unsigned short* __restrict__ wihf,  // [512,256] bf16
    const unsigned short* __restrict__ whhf,  // [512,128] bf16
    const float*          __restrict__ bf_,   // [512]
    const unsigned short* __restrict__ wihb,
    const unsigned short* __restrict__ whhb,
    const float*          __restrict__ bb_,
    const int* __restrict__ len_q,
    const int* __restrict__ len_qp,
    float* __restrict__ embq,                 // [B,256] (fwd cols 0..127, bwd 128..255)
    float* __restrict__ embqp)
{
  const int y   = blockIdx.y;
  const int dir = y & 1;
  const unsigned short* x_bf = (y < 2) ? xq   : xqp;
  const unsigned short* w_ih = dir ? wihb : wihf;
  const unsigned short* w_hh = dir ? whhb : whhf;
  const float*          bias = dir ? bb_  : bf_;
  const int*            lens = (y < 2) ? len_q : len_qp;
  float*                out  = (y < 2) ? embq  : embqp;
  const int colOff = dir * H_;

  __shared__ unsigned short xs [16 * (D_ + 8)];   // staged x tile (padded rows)
  __shared__ unsigned short hsm[16 * (H_ + 8)];   // hidden state, bf16
  __shared__ float          zs [16 * NG_];        // gate pre-activations, f32

  const int tid  = threadIdx.x;
  const int lane = tid & 31;
  const int wv   = tid >> 5;
  const int arow = lane & 15;       // A-fragment row / C-fragment column
  const int ksel = lane >> 4;       // K-half select (contiguous: k = 16*ksel + s)
  const int b0   = blockIdx.x * 16;

  for (int i = tid; i < 16 * (H_ + 8); i += 256) hsm[i] = 0;

  // per-thread cell state: row = tid>>4, 8 consecutive h-columns
  float cst[8];
  #pragma unroll
  for (int u = 0; u < 8; ++u) cst[u] = 0.f;
  const int crow  = tid >> 4;
  const int ccol0 = (tid & 15) * 8;
  const int mylen = lens[b0 + crow];
  // fwd: capture h at step len-1 ; bwd (reversed scan): at step T-len.
  const int capS = (dir == 0) ? (mylen - 1) : (T_ - mylen);

  // Preload loop-invariant W_hh B-fragments and bias into registers.
  v16u  whhF[4][4];
  float bv4[4];
  const unsigned short* wiRow[4];
  #pragma unroll
  for (int j = 0; j < 4; ++j) {
    const int n = (wv * 4 + j) * 16 + arow;
    bv4[j]   = bias[n];
    wiRow[j] = w_ih + (size_t)n * D_;
    const unsigned short* wh = w_hh + (size_t)n * H_;
    #pragma unroll
    for (int kt = 0; kt < 4; ++kt) {
      const unsigned short* p = wh + kt * 32 + ksel * 16;
      whhF[j][kt] = ld2x16(p, p + 8);
    }
  }

  // x staging assignment: each thread moves 16 contiguous halves
  const size_t xRowStride = (size_t)T_ * D_;
  const int myXr = (tid * 16) >> 8;
  const int myXd = (tid * 16) & 255;
  const unsigned short* xBase = x_bf + (size_t)(b0 + myXr) * xRowStride + myXd;

  __syncthreads();

  for (int s = 0; s < T_; ++s) {
    const int t = dir ? (T_ - 1 - s) : s;

    { // stage x[16, 256] tile into LDS
      const unsigned short* src = xBase + (size_t)t * D_;
      u32x4 v0 = ((const u32x4*)src)[0];
      u32x4 v1 = ((const u32x4*)src)[1];
      *(u32x4*)&xs[myXr * (D_ + 8) + myXd]     = v0;
      *(u32x4*)&xs[myXr * (D_ + 8) + myXd + 8] = v1;
    }
    __syncthreads();

    // z = bias + x_t @ W_ih^T + h @ W_hh^T   (per-wave: 4 N-tiles × 12 WMMAs)
    #pragma unroll
    for (int j = 0; j < 4; ++j) {
      const int n = (wv * 4 + j) * 16 + arow;
      const float bv = bv4[j];
      v8f acc = {bv, bv, bv, bv, bv, bv, bv, bv};
      const unsigned short* wi = wiRow[j];
      const unsigned short* ax = &xs[arow * (D_ + 8)];
      #pragma unroll
      for (int kt = 0; kt < 8; ++kt) {
        // A frag: halves 0..15 at k0..k0+15 (contiguous, matches B)
        const int k0 = ksel * 16 + kt * 32;
        v16u a = ld2x16(ax + k0, ax + k0 + 8);
        const unsigned short* p = wi + kt * 32 + ksel * 16;
        v16u b = ld2x16(p, p + 8);
        acc = wmma_bf(a, b, acc);
      }
      const unsigned short* ah = &hsm[arow * (H_ + 8)];
      #pragma unroll
      for (int kt = 0; kt < 4; ++kt) {
        const int k0 = ksel * 16 + kt * 32;
        v16u a = ld2x16(ah + k0, ah + k0 + 8);
        acc = wmma_bf(a, whhF[j][kt], acc);
      }
      // C/D layout: m = vgpr + 8*ksel, col = n
      #pragma unroll
      for (int v = 0; v < 8; ++v)
        zs[(v + ksel * 8) * NG_ + n] = acc[v];
    }
    __syncthreads();

    // LSTM cell (torch gate order i,f,g,o), f32 state
    #pragma unroll
    for (int u = 0; u < 8; ++u) {
      const int col = ccol0 + u;
      const float zi = zs[crow * NG_ + col];
      const float zf = zs[crow * NG_ + H_ + col];
      const float zg = zs[crow * NG_ + 2 * H_ + col];
      const float zo = zs[crow * NG_ + 3 * H_ + col];
      cst[u] = sigm(zf) * cst[u] + sigm(zi) * tanhf(zg);
      const float h = sigm(zo) * tanhf(cst[u]);
      hsm[crow * (H_ + 8) + col] = f2bf(h);
      if (s == capS)
        out[(size_t)(b0 + crow) * (2 * H_) + colOff + col] = h;
    }
    __syncthreads();
  }
}

// ---------- conv/FC head: one block per sample, one channel per thread ----------
__device__ __forceinline__ float dot256(const float* __restrict__ a, const float* __restrict__ b) {
  float s = 0.f;
  #pragma unroll 8
  for (int d = 0; d < 256; ++d) s = fmaf(a[d], b[d], s);
  return s;
}

__global__ __launch_bounds__(128) void head_kernel(
    const float* __restrict__ ru_emb, const float* __restrict__ au_emb,
    const float* __restrict__ embq,   const float* __restrict__ embqp,
    const int* __restrict__ r_idx, const int* __restrict__ a_idx,
    const int* __restrict__ acc_idx, const int* __restrict__ rp_idx,
    const int* __restrict__ ap_idx,
    const float* __restrict__ c1w, const float* __restrict__ c1b,
    const float* __restrict__ c2w, const float* __restrict__ c2b,
    const float* __restrict__ c3w, const float* __restrict__ c3b,
    const float* __restrict__ p1w, const float* __restrict__ p1b,
    const float* __restrict__ p2w, const float* __restrict__ p2b,
    const float* __restrict__ p3w, const float* __restrict__ p3b,
    const float* __restrict__ fc1w, const float* __restrict__ fc1b,
    const float* __restrict__ fc2w, const float* __restrict__ fc2b,
    const float* __restrict__ fp1w, const float* __restrict__ fp1b,
    const float* __restrict__ fp2w, const float* __restrict__ fp2b,
    const float* __restrict__ point, float* __restrict__ partial)
{
  __shared__ float rows[7][256];   // r, q, a, acc, rp, qp, ap
  __shared__ float red[3][128];
  const int b = blockIdx.x;
  const int c = threadIdx.x;

  const float* src0 = ru_emb + (size_t)r_idx[b]   * D_;
  const float* src1 = embq   + (size_t)b          * D_;
  const float* src2 = au_emb + (size_t)a_idx[b]   * D_;
  const float* src3 = au_emb + (size_t)acc_idx[b] * D_;
  const float* src4 = ru_emb + (size_t)rp_idx[b]  * D_;
  const float* src5 = embqp  + (size_t)b          * D_;
  const float* src6 = ru_emb + (size_t)ap_idx[b]  * D_;   // note: ru_emb per reference
  for (int d = c; d < 256; d += 128) {
    rows[0][d] = src0[d]; rows[1][d] = src1[d]; rows[2][d] = src2[d];
    rows[3][d] = src3[d]; rows[4][d] = src4[d]; rows[5][d] = src5[d];
    rows[6][d] = src6[d];
  }
  __syncthreads();

  // ----- regular conv set, channel c -----
  const float* w1  = c1w + (size_t)c * D_;
  const float* w20 = c2w + (size_t)c * 2 * D_;
  const float* w21 = w20 + D_;
  const float* w30 = c3w + (size_t)c * 3 * D_;
  const float* w31 = w30 + D_;
  const float* w32 = w30 + 2 * D_;
  const float b1 = c1b[c], b2 = c2b[c], b3 = c3b[c];

  float d1r = dot256(rows[0], w1), d1q = dot256(rows[1], w1);
  float d1a = dot256(rows[2], w1), d1c = dot256(rows[3], w1);
  float d20r = dot256(rows[0], w20), d20q = dot256(rows[1], w20);
  float d21q = dot256(rows[1], w21), d21a = dot256(rows[2], w21), d21c = dot256(rows[3], w21);
  float d30r = dot256(rows[0], w30);
  float d31q = dot256(rows[1], w31);
  float d32a = dot256(rows[2], w32), d32c = dot256(rows[3], w32);

  const float sL0 = fmaxf(d1r + b1, 0.f);
  const float sL1 = fmaxf(d1q + b1, 0.f);
  const float sL2 = fmaxf(d1a + b1, 0.f);
  const float sH2 = fmaxf(d1c + b1, 0.f);
  const float sL3 = fmaxf(d20r + d21q + b2, 0.f);
  const float sL4 = fmaxf(d20q + d21a + b2, 0.f);
  const float sH4 = fmaxf(d20q + d21c + b2, 0.f);
  const float sL5 = fmaxf(d30r + d31q + d32a + b3, 0.f);
  const float sH5 = fmaxf(d30r + d31q + d32c + b3, 0.f);

  const float s1L = fc1b[0] + sL0*fc1w[0] + sL1*fc1w[1] + sL2*fc1w[2]
                            + sL3*fc1w[3] + sL4*fc1w[4] + sL5*fc1w[5];
  const float s1H = fc1b[0] + sL0*fc1w[0] + sL1*fc1w[1] + sH2*fc1w[2]
                            + sL3*fc1w[3] + sH4*fc1w[4] + sH5*fc1w[5];

  // ----- point conv set (rows rp, qp, ap) -----
  const float* q1  = p1w + (size_t)c * D_;
  const float* q20 = p2w + (size_t)c * 2 * D_;
  const float* q21 = q20 + D_;
  const float* q30 = p3w + (size_t)c * 3 * D_;
  const float* q31 = q30 + D_;
  const float* q32 = q30 + 2 * D_;
  const float pb1 = p1b[c], pb2 = p2b[c], pb3 = p3b[c];

  float e10 = dot256(rows[4], q1), e11 = dot256(rows[5], q1), e12 = dot256(rows[6], q1);
  float e200 = dot256(rows[4], q20), e201 = dot256(rows[5], q20);
  float e211 = dot256(rows[5], q21), e212 = dot256(rows[6], q21);
  float e30 = dot256(rows[4], q30), e31 = dot256(rows[5], q31), e32 = dot256(rows[6], q32);

  const float sP0 = fmaxf(e10 + pb1, 0.f);
  const float sP1 = fmaxf(e11 + pb1, 0.f);
  const float sP2 = fmaxf(e12 + pb1, 0.f);
  const float sP3 = fmaxf(e200 + e211 + pb2, 0.f);
  const float sP4 = fmaxf(e201 + e212 + pb2, 0.f);
  const float sP5 = fmaxf(e30 + e31 + e32 + pb3, 0.f);
  const float s1P = fp1b[0] + sP0*fp1w[0] + sP1*fp1w[1] + sP2*fp1w[2]
                            + sP3*fp1w[3] + sP4*fp1w[4] + sP5*fp1w[5];

  red[0][c] = s1L * fc2w[c];
  red[1][c] = s1H * fc2w[c];
  red[2][c] = s1P * fp2w[c];
  __syncthreads();
  for (int offr = 64; offr > 0; offr >>= 1) {
    if (c < offr) {
      red[0][c] += red[0][c + offr];
      red[1][c] += red[1][c + offr];
      red[2][c] += red[2][c + offr];
    }
    __syncthreads();
  }
  if (c == 0) {
    const float low1  = red[0][0] + fc2b[0];
    const float high1 = red[1][0] + fc2b[0];
    const float pts   = red[2][0] + fp2b[0];
    const float pl = sigm(fabsf(pts - point[b]));
    const float rl = sigm(low1 - high1);
    partial[b] = 0.8f * rl + 0.2f * pl;
  }
}

__global__ __launch_bounds__(256) void reduce_kernel(const float* __restrict__ partial,
                                                     float* __restrict__ out, int n) {
  __shared__ float red[256];
  float a = 0.f;
  for (int i = threadIdx.x; i < n; i += 256) a += partial[i];
  red[threadIdx.x] = a;
  __syncthreads();
  for (int off = 128; off > 0; off >>= 1) {
    if (threadIdx.x < off) red[threadIdx.x] += red[threadIdx.x + off];
    __syncthreads();
  }
  if (threadIdx.x == 0) out[0] = red[0];
}

// ---------- launcher ----------
extern "C" void kernel_launch(void* const* d_in, const int* in_sizes, int n_in,
                              void* d_out, int out_size, void* d_ws, size_t ws_size,
                              hipStream_t stream) {
  (void)in_sizes; (void)n_in; (void)out_size; (void)ws_size;

  const float* ru_emb = (const float*)d_in[0];
  const float* au_emb = (const float*)d_in[1];
  const float* w_ih_f = (const float*)d_in[2];
  const float* w_hh_f = (const float*)d_in[3];
  const float* b_f    = (const float*)d_in[4];
  const float* w_ih_b = (const float*)d_in[5];
  const float* w_hh_b = (const float*)d_in[6];
  const float* b_b    = (const float*)d_in[7];
  const float* c1w = (const float*)d_in[8];
  const float* c1b = (const float*)d_in[9];
  const float* c2w = (const float*)d_in[10];
  const float* c2b = (const float*)d_in[11];
  const float* c3w = (const float*)d_in[12];
  const float* c3b = (const float*)d_in[13];
  const float* p1w = (const float*)d_in[14];
  const float* p1b = (const float*)d_in[15];
  const float* p2w = (const float*)d_in[16];
  const float* p2b = (const float*)d_in[17];
  const float* p3w = (const float*)d_in[18];
  const float* p3b = (const float*)d_in[19];
  const float* fc1w = (const float*)d_in[20];
  const float* fc1b = (const float*)d_in[21];
  const float* fc2w = (const float*)d_in[22];
  const float* fc2b = (const float*)d_in[23];
  const float* fp1w = (const float*)d_in[24];
  const float* fp1b = (const float*)d_in[25];
  const float* fp2w = (const float*)d_in[26];
  const float* fp2b = (const float*)d_in[27];
  const int*   r_idx   = (const int*)d_in[28];
  const int*   a_idx   = (const int*)d_in[29];
  const int*   acc_idx = (const int*)d_in[30];
  const float* rank_q  = (const float*)d_in[31];
  const int*   q_len   = (const int*)d_in[32];
  const int*   rp_idx  = (const int*)d_in[33];
  const int*   ap_idx  = (const int*)d_in[34];
  const float* rank_qp = (const float*)d_in[35];
  const int*   qp_len  = (const int*)d_in[36];
  const float* point   = (const float*)d_in[37];

  // ---- workspace carve-up (256B aligned) ----
  char* base = (char*)d_ws;
  size_t off = 0;
  auto carve = [&](size_t bytes) -> void* {
    void* p = base + off;
    off += (bytes + 255) & ~(size_t)255;
    return p;
  };
  const size_t nX = (size_t)B_ * T_ * D_;
  unsigned short* xq   = (unsigned short*)carve(nX * 2);
  unsigned short* xqp  = (unsigned short*)carve(nX * 2);
  unsigned short* wihf = (unsigned short*)carve((size_t)NG_ * D_ * 2);
  unsigned short* wihb = (unsigned short*)carve((size_t)NG_ * D_ * 2);
  unsigned short* whhf = (unsigned short*)carve((size_t)NG_ * H_ * 2);
  unsigned short* whhb = (unsigned short*)carve((size_t)NG_ * H_ * 2);
  float* embq  = (float*)carve((size_t)B_ * D_ * 4);
  float* embqp = (float*)carve((size_t)B_ * D_ * 4);
  float* part  = (float*)carve((size_t)B_ * 4);

  // 1) bf16 conversions
  cvt_bf16<<<2048, 256, 0, stream>>>(rank_q,  xq,  (int)nX);
  cvt_bf16<<<2048, 256, 0, stream>>>(rank_qp, xqp, (int)nX);
  cvt_bf16<<<256, 256, 0, stream>>>(w_ih_f, wihf, NG_ * D_);
  cvt_bf16<<<256, 256, 0, stream>>>(w_ih_b, wihb, NG_ * D_);
  cvt_bf16<<<128, 256, 0, stream>>>(w_hh_f, whhf, NG_ * H_);
  cvt_bf16<<<128, 256, 0, stream>>>(w_hh_b, whhb, NG_ * H_);

  // 2) zero gather outputs (len==0 rows must stay zero)
  zero_f32<<<256, 256, 0, stream>>>(embq,  B_ * D_);
  zero_f32<<<256, 256, 0, stream>>>(embqp, B_ * D_);

  // 3) fused BiLSTM recurrence with WMMA (both sequences, both directions)
  dim3 grec(B_ / 16, 4);
  bilstm_rec<<<grec, 256, 0, stream>>>(xq, xqp, wihf, whhf, b_f, wihb, whhb, b_b,
                                       q_len, qp_len, embq, embqp);

  // 4) conv/FC head per sample
  head_kernel<<<B_, 128, 0, stream>>>(ru_emb, au_emb, embq, embqp,
                                      r_idx, a_idx, acc_idx, rp_idx, ap_idx,
                                      c1w, c1b, c2w, c2b, c3w, c3b,
                                      p1w, p1b, p2w, p2b, p3w, p3b,
                                      fc1w, fc1b, fc2w, fc2b,
                                      fp1w, fp1b, fp2w, fp2b,
                                      point, part);

  // 5) deterministic final reduction -> scalar loss
  reduce_kernel<<<1, 256, 0, stream>>>(part, (float*)d_out, B_);
}
